// CrossAttentionwithAttInv_25872882991667
// MI455X (gfx1250) — matrix operand order repaired and checked by reading, not compiled, not run on hardware
//
#include <hip/hip_runtime.h>

#define Bb 8
#define Nn 1024
#define Cc 768
#define Hh 12
#define Dd 64
#define Mm (Bb * Nn)      // 8192 tokens
#define ATT_SCALE 0.125f  // 64^-0.5
#define KSTEPS (Cc / 32)  // 24

typedef __attribute__((ext_vector_type(16))) __bf16 v16bf;
typedef __attribute__((ext_vector_type(8)))  float  v8f;
typedef int i4v __attribute__((vector_size(16)));   // matches async-LDS builtin param

struct alignas(16) U4 { unsigned int x, y, z, w; };
union Frag16 { U4 q[2]; v16bf bf; };

// ---- optional CDNA5 async global->LDS path (ASYNCcnt), guarded so the file
// ---- always compiles; fallback is global->VGPR->ds_store.
#if defined(__has_builtin)
#if __has_builtin(__builtin_amdgcn_global_load_async_to_lds_b128) && \
    __has_builtin(__builtin_amdgcn_s_wait_asynccnt)
#define USE_ASYNC_LDS 1
#endif
#endif
#ifndef USE_ASYNC_LDS
#define USE_ASYNC_LDS 0
#endif

#define AS1 __attribute__((address_space(1)))
#define AS3 __attribute__((address_space(3)))

__device__ __forceinline__ void copy16_g2l(unsigned short* l, const unsigned short* g) {
#if USE_ASYNC_LDS
  __builtin_amdgcn_global_load_async_to_lds_b128((AS1 i4v*)(g), (AS3 i4v*)(l), 0, 0);
#else
  *(U4*)l = *(const U4*)g;
#endif
}
__device__ __forceinline__ void tile_fence() {
#if USE_ASYNC_LDS
  __builtin_amdgcn_s_wait_asynccnt(0);
#endif
  __syncthreads();
}

__device__ __forceinline__ unsigned short f2bf(float x) {
  unsigned int u = __float_as_uint(x);
  u += 0x7FFFu + ((u >> 16) & 1u);   // round to nearest even
  return (unsigned short)(u >> 16);
}
__device__ __forceinline__ float bf2f(unsigned short h) {
  return __uint_as_float(((unsigned int)h) << 16);
}

__device__ __forceinline__ v8f wmma_bf16(const Frag16& a, const Frag16& b, v8f c) {
  return __builtin_amdgcn_wmma_f32_16x16x32_bf16(false, a.bf, false, b.bf,
                                                 (short)0, c, false, false);
}

// ---------------------------------------------------------------------------
// Kernel 1: fp32 -> bf16 weight conversion (Wq, Wk, Wv, Wp)
// ---------------------------------------------------------------------------
__global__ void cvt_weights_k(const float* __restrict__ w0, const float* __restrict__ w1,
                              const float* __restrict__ w2, const float* __restrict__ w3,
                              unsigned short* __restrict__ o0, unsigned short* __restrict__ o1,
                              unsigned short* __restrict__ o2, unsigned short* __restrict__ o3) {
  int i = blockIdx.x * 256 + threadIdx.x;
  if (i < Cc * Cc) {
    o0[i] = f2bf(w0[i]); o1[i] = f2bf(w1[i]);
    o2[i] = f2bf(w2[i]); o3[i] = f2bf(w3[i]);
  }
}

// ---------------------------------------------------------------------------
// Kernel 2: per-token prep: bf16 casts of q_in/kv_in, StarReLU feature,
// tiny gate linears (H=12 outputs each) via wave reductions.
// ---------------------------------------------------------------------------
__global__ __launch_bounds__(256) void prep_k(
    const float* __restrict__ q_in, const float* __restrict__ kv_in,
    const float* __restrict__ W_dy2, const float* __restrict__ b_dy2,
    const float* __restrict__ W_dy, const float* __restrict__ b_dy,
    const float* __restrict__ star_scale, const float* __restrict__ star_bias,
    unsigned short* __restrict__ qb, unsigned short* __restrict__ kvb,
    float* __restrict__ dy_lf, float* __restrict__ dy_hf) {
  __shared__ float feat[Cc];
  const int tok = blockIdx.x;
  const float ss = star_scale[0], sb = star_bias[0];
  const float* qr = q_in + (size_t)tok * Cc;
  const float* kr = kv_in + (size_t)tok * Cc;
  for (int i = threadIdx.x; i < Cc; i += 256) {
    float x = qr[i];
    qb[(size_t)tok * Cc + i]  = f2bf(x);
    kvb[(size_t)tok * Cc + i] = f2bf(kr[i]);
    float r = fmaxf(x, 0.0f);
    feat[i] = ss * r * r + sb;  // StarReLU
  }
  __syncthreads();
  const int wv = threadIdx.x >> 5, ln = threadIdx.x & 31;
  for (int o = wv; o < 2 * Hh; o += 8) {
    const float* wr = (o < Hh) ? (W_dy2 + (size_t)o * Cc) : (W_dy + (size_t)(o - Hh) * Cc);
    float s = 0.0f;
    for (int j = ln; j < Cc; j += 32) s += feat[j] * wr[j];
    s += __shfl_xor(s, 16);
    s += __shfl_xor(s, 8);
    s += __shfl_xor(s, 4);
    s += __shfl_xor(s, 2);
    s += __shfl_xor(s, 1);
    if (ln == 0) {
      if (o < Hh) {
        dy_lf[(size_t)tok * Hh + o] = tanhf(s + b_dy2[o]);
      } else {
        float x = s + b_dy[o - Hh];
        float sp = (x > 20.0f) ? x : log1pf(__expf(x));   // softplus
        float s2 = sp * sp;
        dy_hf[(size_t)tok * Hh + (o - Hh)] = 2.0f * s2 / (s2 + 0.3678f);
      }
    }
  }
}

// ---------------------------------------------------------------------------
// Generic GEMM: Y[m][n] = sum_c A[m][c] * W[n][c]
// Block tile 128(M) x 128(N), K-step 32, double-buffered LDS (async staging
// when available). 8 waves as 4(M) x 2(N); each wave 32x64 = 8 WMMA accums.
// Output modes: 0: bf16 (B,H,N,D)   1: bf16 (B,H,D,N)   2: fp32 + bias
// ---------------------------------------------------------------------------
__device__ __forceinline__ void gemm_store_tile(v8f c, int mode, void* dst,
                                                const float* bias, int gm, int col,
                                                int hi) {
  if (mode == 2) {
    float* out = (float*)dst;
    float bv = bias[col];
#pragma unroll
    for (int r = 0; r < 8; ++r) {
      int row = gm + r + 8 * hi;
      out[(size_t)row * Cc + col] = c[r] + bv;
    }
  } else {
    unsigned short* out = (unsigned short*)dst;
    int h = col >> 6, d = col & 63;
#pragma unroll
    for (int r = 0; r < 8; ++r) {
      int row = gm + r + 8 * hi;
      int b = row >> 10, n = row & 1023;
      size_t idx = (mode == 0)
          ? ((size_t)((b * Hh + h) * Nn + n) * Dd + d)    // (B,H,N,D)
          : ((size_t)((b * Hh + h) * Dd + d) * Nn + n);   // (B,H,D,N)
      out[idx] = f2bf(c[r]);
    }
  }
}

__global__ __launch_bounds__(256) void gemm_bf16_k(
    const unsigned short* __restrict__ A, const unsigned short* __restrict__ W,
    void* __restrict__ dst, const float* __restrict__ bias, int mode) {
  // stride 40 shorts (80B): rows 16B-aligned, banks spread
  __shared__ __align__(16) unsigned short As[2][128 * 40];
  __shared__ __align__(16) unsigned short Bs[2][128 * 40];
  const int t = threadIdx.x;
  const int wv = t >> 5, ln = t & 31;
  const int lm = ln & 15, hi = ln >> 4;
  const int wvm = wv & 3, wvn = wv >> 2;
  const int bm = blockIdx.y * 128;
  const int bn = blockIdx.x * 128;

  // per-thread staging slots: 2 A chunks + 2 B chunks per K-step
  const int m0 = t >> 2, p0 = t & 3;   // rows 0..63
  const int m1 = m0 + 64;              // rows 64..127

  auto issue = [&](int buf, int kk) {
    copy16_g2l(&As[buf][m0 * 40 + p0 * 8], A + (size_t)(bm + m0) * Cc + kk + p0 * 8);
    copy16_g2l(&As[buf][m1 * 40 + p0 * 8], A + (size_t)(bm + m1) * Cc + kk + p0 * 8);
    copy16_g2l(&Bs[buf][m0 * 40 + p0 * 8], W + (size_t)(bn + m0) * Cc + kk + p0 * 8);
    copy16_g2l(&Bs[buf][m1 * 40 + p0 * 8], W + (size_t)(bn + m1) * Cc + kk + p0 * 8);
  };

  v8f c00 = {}, c01 = {}, c02 = {}, c03 = {};
  v8f c10 = {}, c11 = {}, c12 = {}, c13 = {};

  auto compute = [&](int buf) {
    const unsigned short* as_ = &As[buf][0];
    const unsigned short* bs_ = &Bs[buf][0];
    Frag16 fa0, fa1, fb;
    // A 16x32 frag: lane half hi holds K = hi*8..+7 and 16+hi*8..+7
    fa0.q[0] = *(const U4*)(as_ + (wvm * 32 + lm) * 40 + hi * 8);
    fa0.q[1] = *(const U4*)(as_ + (wvm * 32 + lm) * 40 + 16 + hi * 8);
    fa1.q[0] = *(const U4*)(as_ + (wvm * 32 + 16 + lm) * 40 + hi * 8);
    fa1.q[1] = *(const U4*)(as_ + (wvm * 32 + 16 + lm) * 40 + 16 + hi * 8);
    // B 32x16 frag: lane = column, K = hi*8..+7 and 16+hi*8..+7
    fb.q[0] = *(const U4*)(bs_ + (wvn * 64 + 0 * 16 + lm) * 40 + hi * 8);
    fb.q[1] = *(const U4*)(bs_ + (wvn * 64 + 0 * 16 + lm) * 40 + 16 + hi * 8);
    c00 = wmma_bf16(fa0, fb, c00);
    c10 = wmma_bf16(fa1, fb, c10);
    fb.q[0] = *(const U4*)(bs_ + (wvn * 64 + 1 * 16 + lm) * 40 + hi * 8);
    fb.q[1] = *(const U4*)(bs_ + (wvn * 64 + 1 * 16 + lm) * 40 + 16 + hi * 8);
    c01 = wmma_bf16(fa0, fb, c01);
    c11 = wmma_bf16(fa1, fb, c11);
    fb.q[0] = *(const U4*)(bs_ + (wvn * 64 + 2 * 16 + lm) * 40 + hi * 8);
    fb.q[1] = *(const U4*)(bs_ + (wvn * 64 + 2 * 16 + lm) * 40 + 16 + hi * 8);
    c02 = wmma_bf16(fa0, fb, c02);
    c12 = wmma_bf16(fa1, fb, c12);
    fb.q[0] = *(const U4*)(bs_ + (wvn * 64 + 3 * 16 + lm) * 40 + hi * 8);
    fb.q[1] = *(const U4*)(bs_ + (wvn * 64 + 3 * 16 + lm) * 40 + 16 + hi * 8);
    c03 = wmma_bf16(fa0, fb, c03);
    c13 = wmma_bf16(fa1, fb, c13);
  };

  issue(0, 0);
  for (int s = 0; s < KSTEPS; ++s) {
    tile_fence();                       // tile s resident; prior readers drained
    if (s + 1 < KSTEPS) issue((s + 1) & 1, (s + 1) * 32);
    compute(s & 1);
  }

  const int gm0 = bm + wvm * 32;
  const int gm1 = gm0 + 16;
  const int cb = bn + wvn * 64;
  gemm_store_tile(c00, mode, dst, bias, gm0, cb + 0,  hi);
  gemm_store_tile(c01, mode, dst, bias, gm0, cb + 16, hi);
  gemm_store_tile(c02, mode, dst, bias, gm0, cb + 32, hi);
  gemm_store_tile(c03, mode, dst, bias, gm0, cb + 48, hi);
  gemm_store_tile(c10, mode, dst, bias, gm1, cb + 0,  hi);
  gemm_store_tile(c11, mode, dst, bias, gm1, cb + 16, hi);
  gemm_store_tile(c12, mode, dst, bias, gm1, cb + 32, hi);
  gemm_store_tile(c13, mode, dst, bias, gm1, cb + 48, hi);
}

// ---------------------------------------------------------------------------
// Kernel 4: flash attention + fused gating epilogue.
// grid (N/128, B*H); block 256 = 8 waves; each wave: 16 queries x D=64.
// Q,K: (B,H,N,D) bf16.  Vt: (B,H,D,N) bf16.  Writes gated x to xg (B,N,C) bf16.
// ---------------------------------------------------------------------------
__device__ __forceinline__ void attn_gate_store(
    v8f o, int j, int b, int h, int qw, int hi, int lm,
    const unsigned short* Vp, const float* __restrict__ lf_gamma,
    const float* __restrict__ hf_gamma, const float* dlf, const float* dhf,
    const float* inv, unsigned short* __restrict__ xg) {
  int dloc = j * 16 + lm;
  int c = h * Dd + dloc;
  float lg = lf_gamma[c], hg = hf_gamma[c];
  union { U4 r4; unsigned short us[8]; } vr;
  vr.r4 = *(const U4*)(Vp + (size_t)dloc * Nn + qw + hi * 8);
#pragma unroll
  for (int r = 0; r < 8; ++r) {
    int qi = qw + r + 8 * hi;
    float x = o[r] * inv[r];                 // softmax-normalized attention out
    float vv = bf2f(vr.us[r]);               // v value at this token (Nk==Nq)
    x = x + x * dlf[r] * lg + dhf[r] * (vv - x) * hg;
    xg[(size_t)(b * Nn + qi) * Cc + c] = f2bf(x);
  }
}

__global__ __launch_bounds__(256) void attn_k(
    const unsigned short* __restrict__ Q, const unsigned short* __restrict__ K,
    const unsigned short* __restrict__ Vt,
    const float* __restrict__ dy_lf, const float* __restrict__ dy_hf,
    const float* __restrict__ lf_gamma, const float* __restrict__ hf_gamma,
    unsigned short* __restrict__ xg) {
  __shared__ __align__(16) unsigned short Plds[8][16 * 32];  // per-wave P slab
  const int t = threadIdx.x, wv = t >> 5, ln = t & 31;
  const int lm = ln & 15, hi = ln >> 4;
  const int bh = blockIdx.y;
  const int b = bh / Hh, h = bh % Hh;
  const int qw = blockIdx.x * 128 + wv * 16;

  const unsigned short* Qp = Q + ((size_t)bh * Nn + qw) * Dd;
  const unsigned short* Kp = K + (size_t)bh * Nn * Dd;
  const unsigned short* Vp = Vt + (size_t)bh * Dd * Nn;

  Frag16 a0, a1;  // Q 16x64 as two A fragments (d 0-31, d 32-63)
  a0.q[0] = *(const U4*)(Qp + lm * Dd + hi * 8);
  a0.q[1] = *(const U4*)(Qp + lm * Dd + 16 + hi * 8);
  a1.q[0] = *(const U4*)(Qp + lm * Dd + 32 + hi * 8);
  a1.q[1] = *(const U4*)(Qp + lm * Dd + 48 + hi * 8);

  v8f o0 = {}, o1 = {}, o2 = {}, o3 = {};
  float mrow[8], lrow[8];
#pragma unroll
  for (int r = 0; r < 8; ++r) { mrow[r] = -3.0e38f; lrow[r] = 0.0f; }

  for (int kb = 0; kb < Nn; kb += 32) {
    v8f s0 = {}, s1 = {};
    Frag16 kf;
    // S tile (16 queries x 32 keys): B frag lane = key column, K dim = d
    kf.q[0] = *(const U4*)(Kp + (size_t)(kb + lm) * Dd + hi * 8);
    kf.q[1] = *(const U4*)(Kp + (size_t)(kb + lm) * Dd + 16 + hi * 8);
    s0 = wmma_bf16(a0, kf, s0);
    kf.q[0] = *(const U4*)(Kp + (size_t)(kb + lm) * Dd + 32 + hi * 8);
    kf.q[1] = *(const U4*)(Kp + (size_t)(kb + lm) * Dd + 48 + hi * 8);
    s0 = wmma_bf16(a1, kf, s0);
    kf.q[0] = *(const U4*)(Kp + (size_t)(kb + 16 + lm) * Dd + hi * 8);
    kf.q[1] = *(const U4*)(Kp + (size_t)(kb + 16 + lm) * Dd + 16 + hi * 8);
    s1 = wmma_bf16(a0, kf, s1);
    kf.q[0] = *(const U4*)(Kp + (size_t)(kb + 16 + lm) * Dd + 32 + hi * 8);
    kf.q[1] = *(const U4*)(Kp + (size_t)(kb + 16 + lm) * Dd + 48 + hi * 8);
    s1 = wmma_bf16(a1, kf, s1);

    // online softmax: per-row (half-wave) max / sum reductions
    float fac[8];
#pragma unroll
    for (int r = 0; r < 8; ++r) {
      float v0 = s0[r] * ATT_SCALE;
      float v1 = s1[r] * ATT_SCALE;
      float tm = fmaxf(v0, v1);
      tm = fmaxf(tm, __shfl_xor(tm, 1));
      tm = fmaxf(tm, __shfl_xor(tm, 2));
      tm = fmaxf(tm, __shfl_xor(tm, 4));
      tm = fmaxf(tm, __shfl_xor(tm, 8));
      float mn = fmaxf(mrow[r], tm);
      float fc = __expf(mrow[r] - mn);
      mrow[r] = mn;
      fac[r] = fc;
      float p0 = __expf(v0 - mn);
      float p1 = __expf(v1 - mn);
      float rs = p0 + p1;
      rs += __shfl_xor(rs, 1);
      rs += __shfl_xor(rs, 2);
      rs += __shfl_xor(rs, 4);
      rs += __shfl_xor(rs, 8);
      lrow[r] = lrow[r] * fc + rs;
      // park P (D-layout) in wave-private LDS for relayout to A fragment
      Plds[wv][(r + 8 * hi) * 32 + lm] = f2bf(p0);
      Plds[wv][(r + 8 * hi) * 32 + 16 + lm] = f2bf(p1);
    }
#pragma unroll
    for (int r = 0; r < 8; ++r) {
      o0[r] *= fac[r]; o1[r] *= fac[r]; o2[r] *= fac[r]; o3[r] *= fac[r];
    }
    __builtin_amdgcn_wave_barrier();  // keep LDS write->read order (in-order DS per wave)

    Frag16 pf;  // P 16x32 in A layout
    pf.q[0] = *(const U4*)(&Plds[wv][lm * 32 + hi * 8]);
    pf.q[1] = *(const U4*)(&Plds[wv][lm * 32 + 16 + hi * 8]);

    Frag16 vf;  // V tile (K=32 keys x N=16 d): keys hi*8..+7 and 16+hi*8..+7 from Vt rows
    vf.q[0] = *(const U4*)(Vp + (size_t)(0 * 16 + lm) * Nn + kb + hi * 8);
    vf.q[1] = *(const U4*)(Vp + (size_t)(0 * 16 + lm) * Nn + kb + 16 + hi * 8);
    o0 = wmma_bf16(pf, vf, o0);
    vf.q[0] = *(const U4*)(Vp + (size_t)(1 * 16 + lm) * Nn + kb + hi * 8);
    vf.q[1] = *(const U4*)(Vp + (size_t)(1 * 16 + lm) * Nn + kb + 16 + hi * 8);
    o1 = wmma_bf16(pf, vf, o1);
    vf.q[0] = *(const U4*)(Vp + (size_t)(2 * 16 + lm) * Nn + kb + hi * 8);
    vf.q[1] = *(const U4*)(Vp + (size_t)(2 * 16 + lm) * Nn + kb + 16 + hi * 8);
    o2 = wmma_bf16(pf, vf, o2);
    vf.q[0] = *(const U4*)(Vp + (size_t)(3 * 16 + lm) * Nn + kb + hi * 8);
    vf.q[1] = *(const U4*)(Vp + (size_t)(3 * 16 + lm) * Nn + kb + 16 + hi * 8);
    o3 = wmma_bf16(pf, vf, o3);
  }

  float dlf[8], dhf[8], inv[8];
#pragma unroll
  for (int r = 0; r < 8; ++r) {
    int qi = qw + r + 8 * hi;
    inv[r] = 1.0f / lrow[r];
    dlf[r] = dy_lf[(size_t)(b * Nn + qi) * Hh + h];
    dhf[r] = dy_hf[(size_t)(b * Nn + qi) * Hh + h];
  }
  attn_gate_store(o0, 0, b, h, qw, hi, lm, Vp, lf_gamma, hf_gamma, dlf, dhf, inv, xg);
  attn_gate_store(o1, 1, b, h, qw, hi, lm, Vp, lf_gamma, hf_gamma, dlf, dhf, inv, xg);
  attn_gate_store(o2, 2, b, h, qw, hi, lm, Vp, lf_gamma, hf_gamma, dlf, dhf, inv, xg);
  attn_gate_store(o3, 3, b, h, qw, hi, lm, Vp, lf_gamma, hf_gamma, dlf, dhf, inv, xg);
}

// ---------------------------------------------------------------------------
extern "C" void kernel_launch(void* const* d_in, const int* in_sizes, int n_in,
                              void* d_out, int out_size, void* d_ws, size_t ws_size,
                              hipStream_t stream) {
  (void)in_sizes; (void)n_in; (void)out_size; (void)ws_size;
  const float* q_in      = (const float*)d_in[0];
  const float* kv_in     = (const float*)d_in[1];
  const float* Wq        = (const float*)d_in[2];
  const float* Wk        = (const float*)d_in[3];
  const float* Wv        = (const float*)d_in[4];
  const float* Wp        = (const float*)d_in[5];
  const float* bp        = (const float*)d_in[6];
  const float* W_dy2     = (const float*)d_in[7];
  const float* b_dy2     = (const float*)d_in[8];
  const float* W_dy      = (const float*)d_in[9];
  const float* b_dy      = (const float*)d_in[10];
  const float* lf_gamma  = (const float*)d_in[11];
  const float* hf_gamma  = (const float*)d_in[12];
  const float* star_scale = (const float*)d_in[13];
  const float* star_bias  = (const float*)d_in[14];

  char* w = (char*)d_ws;
  size_t off = 0;
  auto carve = [&](size_t bytes) -> char* {
    char* p = w + off;
    off += (bytes + 255) & ~(size_t)255;
    return p;
  };
  unsigned short* qb  = (unsigned short*)carve((size_t)Mm * Cc * 2);
  unsigned short* kvb = (unsigned short*)carve((size_t)Mm * Cc * 2);
  unsigned short* Wqb = (unsigned short*)carve((size_t)Cc * Cc * 2);
  unsigned short* Wkb = (unsigned short*)carve((size_t)Cc * Cc * 2);
  unsigned short* Wvb = (unsigned short*)carve((size_t)Cc * Cc * 2);
  unsigned short* Wpb = (unsigned short*)carve((size_t)Cc * Cc * 2);
  unsigned short* Qh  = (unsigned short*)carve((size_t)Mm * Cc * 2);  // (B,H,N,D)
  unsigned short* Kh  = (unsigned short*)carve((size_t)Mm * Cc * 2);  // (B,H,N,D)
  unsigned short* Vth = (unsigned short*)carve((size_t)Mm * Cc * 2);  // (B,H,D,N)
  float* dylf = (float*)carve((size_t)Mm * Hh * 4);
  float* dyhf = (float*)carve((size_t)Mm * Hh * 4);
  unsigned short* xg = (unsigned short*)carve((size_t)Mm * Cc * 2);   // gated attn out

  cvt_weights_k<<<(Cc * Cc + 255) / 256, 256, 0, stream>>>(Wq, Wk, Wv, Wp,
                                                           Wqb, Wkb, Wvb, Wpb);
  prep_k<<<Mm, 256, 0, stream>>>(q_in, kv_in, W_dy2, b_dy2, W_dy, b_dy,
                                 star_scale, star_bias, qb, kvb, dylf, dyhf);
  gemm_bf16_k<<<dim3(Cc / 128, Mm / 128), 256, 0, stream>>>(qb,  Wqb, Qh,  nullptr, 0);
  gemm_bf16_k<<<dim3(Cc / 128, Mm / 128), 256, 0, stream>>>(kvb, Wkb, Kh,  nullptr, 0);
  gemm_bf16_k<<<dim3(Cc / 128, Mm / 128), 256, 0, stream>>>(kvb, Wvb, Vth, nullptr, 1);
  attn_k<<<dim3(Nn / 128, Bb * Hh), 256, 0, stream>>>(Qh, Kh, Vth, dylf, dyhf,
                                                      lf_gamma, hf_gamma, xg);
  gemm_bf16_k<<<dim3(Cc / 128, Mm / 128), 256, 0, stream>>>(xg, Wpb, d_out, bp, 2);
}
